// PointNet2Encoder_16192026706631
// MI455X (gfx1250) — matrix operand order repaired and checked by reading, not compiled, not run on hardware
//
#include <hip/hip_runtime.h>
#include <hip/hip_bf16.h>

// ---------------------------------------------------------------------------
// CDNA5 / gfx1250 PointNet++ encoder.
// GEMMs use v_wmma_f32_16x16x32_bf16 (bf16 in, fp32 accum). All other kernels
// are bandwidth-bound helpers (FPS, ball query, grouping, BN stats, pooling).
// ---------------------------------------------------------------------------

typedef __bf16 bf16_t;
typedef __attribute__((ext_vector_type(8)))  __bf16 v8bf;
typedef __attribute__((ext_vector_type(16))) __bf16 v16bf;
typedef __attribute__((ext_vector_type(8)))  float  v8f;

// ---------------- WMMA fragment helpers (wave32) ----------------
// A (16x32 bf16, MxK): lane 0-15 -> row M=lane, halves K = {k0..k0+7, k0+16..k0+23}
//                      lane 16-31 -> row M=lane-16, K = {k0+8..k0+15, k0+24..k0+31}
__device__ __forceinline__ v16bf load_fragA(const bf16_t* __restrict__ base, int ld,
                                            int row0, int k0, int lane) {
  int r  = row0 + (lane & 15);
  int kh = (lane & 16) ? 8 : 0;
  const bf16_t* p = base + (size_t)r * ld + k0 + kh;
  v8bf lo = *(const v8bf*)p;          // K = k0+kh .. +7      (16B load)
  v8bf hi = *(const v8bf*)(p + 16);   // K = k0+16+kh .. +7   (16B load)
  v16bf out;
#pragma unroll
  for (int i = 0; i < 8; i++) { out[i] = lo[i]; out[8 + i] = hi[i]; }
  return out;
}

// B (32x16 bf16, KxN), B = W^T: lane 0-15 -> col N=lane, K = {k0..k0+7, k0+16..k0+23}
//                               lane 16-31 -> col N=lane-16, K = {k0+8..k0+15, k0+24..k0+31}
__device__ __forceinline__ v16bf load_fragB(const bf16_t* __restrict__ base, int ld,
                                            int col0, int k0, int lane) {
  int n  = col0 + (lane & 15);
  int kk = k0 + ((lane & 16) ? 8 : 0);
  const bf16_t* p = base + (size_t)n * ld + kk;
  v8bf lo = *(const v8bf*)p;
  v8bf hi = *(const v8bf*)(p + 16);
  v16bf out;
#pragma unroll
  for (int i = 0; i < 8; i++) { out[i] = lo[i]; out[8 + i] = hi[i]; }
  return out;
}

__device__ __forceinline__ v8f wmma_bf16(v16bf a, v16bf b, v8f c) {
  return __builtin_amdgcn_wmma_f32_16x16x32_bf16(
      /*neg_a=*/false, a, /*neg_b=*/false, b,
      /*c_mod=*/(short)0, c, /*reuse_a=*/false, /*reuse_b=*/false);
}

// C/D (16x16 f32): lane 0-15 -> N=lane, VGPR i = row M=i; lane 16-31 -> N=lane-16, M=8+i
__device__ __forceinline__ void store_tile(float* __restrict__ Y, int ld,
                                           int m0, int n0, int lane, v8f c) {
  int n = n0 + (lane & 15);
  int m = m0 + ((lane & 16) ? 8 : 0);
#pragma unroll
  for (int i = 0; i < 8; i++) Y[(size_t)(m + i) * ld + n] = c[i];
}

// ---------------- GEMM: Y[M,N] = X[M,K] * W[N,K]^T -------------------------
// Grid: (M/128, N/64). Block: 256 threads = 8 waves; wave w covers tile
// (blockIdx.x*128 + (w&3)*32, blockIdx.y*64 + (w>>2)*32), i.e. 32x32 per wave
// via 4 accumulators. M % 128 == 0, N % 64 == 0, K % 32 == 0 guaranteed.
__global__ __launch_bounds__(256) void gemm_bf16_wmma(
    const bf16_t* __restrict__ X, const bf16_t* __restrict__ Wt,
    float* __restrict__ Y, int M, int N, int K) {
  int lane = threadIdx.x & 31;
  int wave = threadIdx.x >> 5;
  int m0 = blockIdx.x * 128 + (wave & 3) * 32;
  int n0 = blockIdx.y * 64 + (wave >> 2) * 32;
  (void)M;
  v8f c00 = {}, c01 = {}, c10 = {}, c11 = {};
  for (int k = 0; k < K; k += 32) {
    v16bf a0 = load_fragA(X, K, m0,      k, lane);
    v16bf a1 = load_fragA(X, K, m0 + 16, k, lane);
    v16bf b0 = load_fragB(Wt, K, n0,      k, lane);
    v16bf b1 = load_fragB(Wt, K, n0 + 16, k, lane);
    c00 = wmma_bf16(a0, b0, c00);
    c01 = wmma_bf16(a0, b1, c01);
    c10 = wmma_bf16(a1, b0, c10);
    c11 = wmma_bf16(a1, b1, c11);
  }
  store_tile(Y, N, m0,      n0,      lane, c00);
  store_tile(Y, N, m0,      n0 + 16, lane, c01);
  store_tile(Y, N, m0 + 16, n0,      lane, c10);
  store_tile(Y, N, m0 + 16, n0 + 16, lane, c11);
}

// ---------------- FPS: one block per cloud ---------------------------------
// Matches reference: sample j is current `far` (far_0 = 0), then dist update,
// then far = argmax(dist) with first-index tie-break (jnp.argmax semantics).
__global__ void fps_kernel(const float* __restrict__ xyz, float* __restrict__ new_xyz,
                           int N, int S) {
  __shared__ float sd[512];
  __shared__ int   si[512];
  int b = blockIdx.x, t = threadIdx.x;
  const float* px = xyz + (size_t)b * N * 3;
  float x = px[t * 3 + 0], y = px[t * 3 + 1], z = px[t * 3 + 2];
  float dist = 1e10f;
  int far = 0;
  for (int j = 0; j < S; j++) {
    float cx = px[far * 3 + 0], cy = px[far * 3 + 1], cz = px[far * 3 + 2];
    if (t == 0) {
      float* o = new_xyz + ((size_t)b * S + j) * 3;
      o[0] = cx; o[1] = cy; o[2] = cz;
    }
    float dx = x - cx, dy = y - cy, dz = z - cz;
    float d = dx * dx + dy * dy + dz * dz;
    dist = fminf(dist, d);
    sd[t] = dist; si[t] = t;
    __syncthreads();
    for (int off = N >> 1; off > 0; off >>= 1) {
      if (t < off) {
        float do_ = sd[t + off], dc = sd[t];
        int   io  = si[t + off], ic = si[t];
        if (do_ > dc || (do_ == dc && io < ic)) { sd[t] = do_; si[t] = io; }
      }
      __syncthreads();
    }
    far = si[0];
    __syncthreads();
  }
}

// ---------------- Ball query: one thread per (b,s) -------------------------
__global__ void ball_query_kernel(const float* __restrict__ xyz,
                                  const float* __restrict__ new_xyz,
                                  int* __restrict__ idx,
                                  int Btot, int N, int S, float r2, int K) {
  int g = blockIdx.x * blockDim.x + threadIdx.x;
  if (g >= Btot * S) return;
  int b = g / S;
  const float* q = new_xyz + (size_t)g * 3;
  float qx = q[0], qy = q[1], qz = q[2];
  const float* px = xyz + (size_t)b * N * 3;
  int* out = idx + (size_t)g * K;
  int cnt = 0;
  for (int i = 0; i < N && cnt < K; i++) {
    float dx = px[i * 3 + 0] - qx;
    float dy = px[i * 3 + 1] - qy;
    float dz = px[i * 3 + 2] - qz;
    if (dx * dx + dy * dy + dz * dz <= r2) out[cnt++] = i;
  }
  int f = (cnt > 0) ? out[0] : 0;
  for (int j = cnt; j < K; j++) out[j] = f;
}

// ---------------- Grouping: gather + center + concat -> bf16 padded --------
__global__ void group_kernel(const float* __restrict__ xyz,
                             const float* __restrict__ new_xyz,
                             const float* __restrict__ feat,
                             const int* __restrict__ idx,
                             bf16_t* __restrict__ Xout,
                             int N, int S, int K, int F, int Cpad) {
  int g = blockIdx.x;            // b*S + s
  int b = g / S;
  const int* id = idx + (size_t)g * K;
  const float* nx = new_xyz + (size_t)g * 3;
  bf16_t* row0 = Xout + (size_t)g * K * Cpad;
  int total = K * Cpad;
  for (int e = threadIdx.x; e < total; e += blockDim.x) {
    int k = e / Cpad, c = e % Cpad;
    int pi = id[k];
    float v = 0.f;
    if (c < 3)            v = xyz[((size_t)b * N + pi) * 3 + c] - nx[c];
    else if (c < 3 + F)   v = feat[((size_t)b * N + pi) * (size_t)F + (c - 3)];
    row0[e] = (bf16_t)v;
  }
}

// SA3 group-all: raw concat(xyz, feat) -> bf16 padded, flat over elements
__global__ void group_all_kernel(const float* __restrict__ xyz,
                                 const float* __restrict__ feat,
                                 bf16_t* __restrict__ Xout,
                                 int F, int Cpad, int total) {
  int e = blockIdx.x * blockDim.x + threadIdx.x;
  if (e >= total) return;
  int g = e / Cpad, c = e % Cpad;   // g = b*S + s
  float v = 0.f;
  if (c < 3)          v = xyz[(size_t)g * 3 + c];
  else if (c < 3 + F) v = feat[(size_t)g * F + (c - 3)];
  Xout[e] = (bf16_t)v;
}

// ---------------- BatchNorm helpers ----------------------------------------
__global__ void zero_kernel(float* __restrict__ p, int n) {
  int i = blockIdx.x * blockDim.x + threadIdx.x;
  if (i < n) p[i] = 0.f;
}

// Per-channel sum / sumsq over M rows. Launch: total threads = 64 * N.
__global__ void stats_kernel(const float* __restrict__ Y, float* __restrict__ sums,
                             int M, int N) {
  int gid = blockIdx.x * blockDim.x + threadIdx.x;
  int walkers = (gridDim.x * blockDim.x) / N;
  int col = gid % N;
  float s = 0.f, ss = 0.f;
  for (int r = gid / N; r < M; r += walkers) {
    float v = Y[(size_t)r * N + col];
    s += v; ss += v * v;
  }
  atomicAdd(&sums[col], s);
  atomicAdd(&sums[N + col], ss);
}

__global__ void bnfinal_kernel(const float* __restrict__ sums,
                               const float* __restrict__ gam,
                               const float* __restrict__ bet,
                               float* __restrict__ sc, int N, float invM) {
  int c = blockIdx.x * blockDim.x + threadIdx.x;
  if (c >= N) return;
  float mean = sums[c] * invM;
  float var  = sums[N + c] * invM - mean * mean;   // biased var (jnp.var)
  float rs   = rsqrtf(var + 1e-5f);
  float scale = gam[c] * rs;
  sc[c]     = scale;
  sc[N + c] = bet[c] - mean * scale;
}

// y = relu(Y*scale + shift); optionally write fp32 (may alias Y) and/or bf16.
__global__ void bn_relu_kernel(const float* __restrict__ Y, const float* __restrict__ sc,
                               float* outF, bf16_t* outB, size_t total, int N) {
  size_t i = (size_t)blockIdx.x * blockDim.x + threadIdx.x;
  if (i >= total) return;
  int c = (int)(i % (size_t)N);
  float v = fmaxf(Y[i] * sc[c] + sc[N + c], 0.f);
  if (outF) outF[i] = v;
  if (outB) outB[i] = (bf16_t)v;
}

// ---------------- Max pool over K within each group ------------------------
__global__ void maxpool_kernel(const float* __restrict__ in, float* __restrict__ out,
                               int Kn, int C) {
  int g = blockIdx.x;
  int c = blockIdx.y * blockDim.x + threadIdx.x;
  if (c >= C) return;
  const float* p = in + (size_t)g * Kn * C + c;
  float m = p[0];
  for (int k = 1; k < Kn; k++) m = fmaxf(m, p[(size_t)k * C]);
  out[(size_t)g * C + c] = m;
}

// ---------------- Weight fp32 -> bf16 with K zero-padding ------------------
__global__ void wconv_kernel(const float* __restrict__ W, bf16_t* __restrict__ Wb,
                             int Cout, int Cin, int Cpad) {
  int e = blockIdx.x * blockDim.x + threadIdx.x;
  if (e >= Cout * Cpad) return;
  int o = e / Cpad, c = e % Cpad;
  Wb[e] = (bf16_t)((c < Cin) ? W[(size_t)o * Cin + c] : 0.f);
}

// ===========================================================================
extern "C" void kernel_launch(void* const* d_in, const int* in_sizes, int n_in,
                              void* d_out, int out_size, void* d_ws, size_t ws_size,
                              hipStream_t stream) {
  (void)in_sizes; (void)n_in; (void)out_size; (void)ws_size;

  // Input order: human_points, then params['sa1'|'sa2'|'sa3'] each = 3x(W,g,b)
  const float* xyz0 = (const float*)d_in[0];
  const float* Wp[9]; const float* Gp[9]; const float* Bp[9];
  for (int l = 0; l < 9; l++) {
    Wp[l] = (const float*)d_in[1 + 3 * l + 0];
    Gp[l] = (const float*)d_in[1 + 3 * l + 1];
    Bp[l] = (const float*)d_in[1 + 3 * l + 2];
  }

  static const int Cin[9]  = {3, 64, 64, 131, 128, 128, 259, 256, 512};
  static const int Cpd[9]  = {32, 64, 64, 160, 128, 128, 288, 256, 512};
  static const int Cout[9] = {64, 64, 128, 128, 128, 256, 256, 512, 1024};

  const int Bc = 128, N0 = 512, S1 = 256, S2 = 128, Kn = 32;
  const int M1 = Bc * S1 * Kn;   // 1,048,576
  const int M2 = Bc * S2 * Kn;   //   524,288
  const int M3 = Bc * S2;        //    16,384

  // ---- workspace bump allocator ----
  char* base = (char*)d_ws;
  size_t off = 0;
  auto alloc = [&](size_t bytes) -> char* {
    char* p = base + off;
    off += (bytes + 255) & ~(size_t)255;
    return p;
  };
  float*  Ybuf = (float*)alloc((size_t)M1 * 128 * sizeof(float));      // 512 MB (== M2*256)
  bf16_t* Xa   = (bf16_t*)alloc((size_t)M2 * 160 * sizeof(bf16_t));    // 160 MB
  bf16_t* Xb   = (bf16_t*)alloc((size_t)M1 * 64 * sizeof(bf16_t));     // 128 MB
  float*  nx1  = (float*)alloc((size_t)Bc * S1 * 3 * sizeof(float));
  float*  nx2  = (float*)alloc((size_t)Bc * S2 * 3 * sizeof(float));
  int*    idx1 = (int*)alloc((size_t)Bc * S1 * Kn * sizeof(int));
  int*    idx2 = (int*)alloc((size_t)Bc * S2 * Kn * sizeof(int));
  float*  feat1 = (float*)alloc((size_t)Bc * S1 * 128 * sizeof(float));
  float*  feat2 = (float*)alloc((size_t)Bc * S2 * 256 * sizeof(float));
  float*  sums = (float*)alloc(2 * 1024 * sizeof(float));
  float*  sc   = (float*)alloc(2 * 1024 * sizeof(float));
  bf16_t* Wb[9];
  for (int l = 0; l < 9; l++)
    Wb[l] = (bf16_t*)alloc((size_t)Cout[l] * Cpd[l] * sizeof(bf16_t));

  // ---- weight conversion (bf16, zero-padded K) ----
  for (int l = 0; l < 9; l++) {
    int tot = Cout[l] * Cpd[l];
    wconv_kernel<<<(tot + 255) / 256, 256, 0, stream>>>(Wp[l], Wb[l], Cout[l], Cin[l], Cpd[l]);
  }

  // ---- one MLP layer: GEMM -> global BN stats -> fused BN+ReLU (+cast) ----
  auto run_layer = [&](const bf16_t* X, int l, int M, float* outF, bf16_t* outB) {
    int N = Cout[l], K = Cpd[l];
    dim3 g(M / 128, N / 64);
    gemm_bf16_wmma<<<g, 256, 0, stream>>>(X, Wb[l], Ybuf, M, N, K);
    zero_kernel<<<(2 * N + 255) / 256, 256, 0, stream>>>(sums, 2 * N);
    stats_kernel<<<N / 4, 256, 0, stream>>>(Ybuf, sums, M, N);   // 64 walkers/channel
    bnfinal_kernel<<<(N + 255) / 256, 256, 0, stream>>>(sums, Gp[l], Bp[l], sc, N, 1.0f / (float)M);
    size_t total = (size_t)M * N;
    bn_relu_kernel<<<(unsigned)((total + 255) / 256), 256, 0, stream>>>(Ybuf, sc, outF, outB, total, N);
  };

  // ================= Stage 1: SA1 (N=512 -> S=256, K=32) ===================
  fps_kernel<<<Bc, N0, 0, stream>>>(xyz0, nx1, N0, S1);
  ball_query_kernel<<<(Bc * S1 + 255) / 256, 256, 0, stream>>>(xyz0, nx1, idx1, Bc, N0, S1, 0.2f * 0.2f, Kn);
  group_kernel<<<Bc * S1, 256, 0, stream>>>(xyz0, nx1, nullptr, idx1, Xa, N0, S1, Kn, 0, Cpd[0]);
  run_layer(Xa, 0, M1, nullptr, Xb);
  run_layer(Xb, 1, M1, nullptr, Xa);
  run_layer(Xa, 2, M1, Ybuf, nullptr);                       // normalize in place
  maxpool_kernel<<<dim3(Bc * S1, 1), 256, 0, stream>>>(Ybuf, feat1, Kn, 128);

  // ================= Stage 2: SA2 (N=256 -> S=128, K=32) ===================
  fps_kernel<<<Bc, S1, 0, stream>>>(nx1, nx2, S1, S2);
  ball_query_kernel<<<(Bc * S2 + 255) / 256, 256, 0, stream>>>(nx1, nx2, idx2, Bc, S1, S2, 0.4f * 0.4f, Kn);
  group_kernel<<<Bc * S2, 256, 0, stream>>>(nx1, nx2, feat1, idx2, Xa, S1, S2, Kn, 128, Cpd[3]);
  run_layer(Xa, 3, M2, nullptr, Xb);
  run_layer(Xb, 4, M2, nullptr, Xa);
  run_layer(Xa, 5, M2, Ybuf, nullptr);
  maxpool_kernel<<<dim3(Bc * S2, 1), 256, 0, stream>>>(Ybuf, feat2, Kn, 256);

  // ================= Stage 3: SA3 (group-all over 128 pts) =================
  int tot3 = M3 * Cpd[6];
  group_all_kernel<<<(tot3 + 255) / 256, 256, 0, stream>>>(nx2, feat2, Xa, 256, Cpd[6], tot3);
  run_layer(Xa, 6, M3, nullptr, Xb);
  run_layer(Xb, 7, M3, nullptr, Xa);
  run_layer(Xa, 8, M3, Ybuf, nullptr);
  maxpool_kernel<<<dim3(Bc, (1024 + 255) / 256), 256, 0, stream>>>(Ybuf, (float*)d_out, S2, 1024);
}
